// MemTransformerLMEncoder_35974646071571
// MI455X (gfx1250) — matrix operand order repaired and checked by reading, not compiled, not run on hardware
//
#include <hip/hip_runtime.h>
#include <hip/hip_bf16.h>

// Model constants (from reference)
#define B_    4
#define S_    1024
#define D_    1024
#define H_    16
#define DH_   64
#define HD_   1024
#define DMEM_ 16
#define KV_LD 2048   // kv projection output width = 2*HD

typedef __attribute__((ext_vector_type(16))) __bf16 v16bf;
typedef __attribute__((ext_vector_type(8)))  __bf16 v8bf;
typedef __attribute__((ext_vector_type(8)))  float  v8f;

static __device__ __forceinline__ v8f wmma_bf16(v16bf a, v16bf b, v8f c) {
  // D = A(16x32) * B(32x16) + C, fp32 accumulate
  return __builtin_amdgcn_wmma_f32_16x16x32_bf16(false, a, false, b, (short)0, c, false, false);
}

// ---------------------------------------------------------------------------
// Fragment loaders (wave32).
// A-operand 16x32 bf16 tile from row-major [rows, ld]:
//   lane<16 : row = row0+lane,    K = k0 + 0..15
//   lane>=16: row = row0+lane-16, K = k0 + 16..31
// (contiguous K order, matching load_b_frag)
// ---------------------------------------------------------------------------
static __device__ __forceinline__ v16bf load_a_frag(const __bf16* base, int row0, int ld, int k0) {
  int lane = threadIdx.x & 31;
  const __bf16* p = base + (size_t)(row0 + (lane & 15)) * ld + k0 + ((lane >> 4) << 4);
  v8bf lo = *(const v8bf*)p;
  v8bf hi = *(const v8bf*)(p + 8);
  v16bf f;
#pragma unroll
  for (int i = 0; i < 8; ++i) { f[i] = lo[i]; f[i + 8] = hi[i]; }
  return f;
}

static __device__ __forceinline__ v16bf load_a_frag_clamped(const __bf16* base, int row0, int maxrow,
                                                            int ld, int k0) {
  int lane = threadIdx.x & 31;
  int r = row0 + (lane & 15);
  r = (r < maxrow) ? r : (maxrow - 1);
  const __bf16* p = base + (size_t)r * ld + k0 + ((lane >> 4) << 4);
  v8bf lo = *(const v8bf*)p;
  v8bf hi = *(const v8bf*)(p + 8);
  v16bf f;
#pragma unroll
  for (int i = 0; i < 8; ++i) { f[i] = lo[i]; f[i + 8] = hi[i]; }
  return f;
}

// B-operand 32x16 tile, taken from Bt stored [N, K] row-major (i.e. B transposed):
//   lane<16 : column n0+lane,    K = k0+0..15  (one contiguous 32B load)
//   lane>=16: column n0+lane-16, K = k0+16..31
static __device__ __forceinline__ v16bf load_b_frag(const __bf16* base, int n0, int ld, int k0) {
  int lane = threadIdx.x & 31;
  return *(const v16bf*)(base + (size_t)(n0 + (lane & 15)) * ld + k0 + ((lane >> 4) << 4));
}

// ---------------------------------------------------------------------------
// Weight prep: W[K,N] fp32 -> Wt[N,K] bf16
// ---------------------------------------------------------------------------
__global__ void k_wt_prep(const float* __restrict__ W, __bf16* __restrict__ Wt, int K, int N) {
  size_t i = (size_t)blockIdx.x * blockDim.x + threadIdx.x;
  if (i >= (size_t)K * N) return;
  int n = (int)(i / K);
  int k = (int)(i - (size_t)n * K);
  Wt[i] = (__bf16)W[(size_t)k * N + n];
}

// ---------------------------------------------------------------------------
// Embedding: X0[b,s,d] = emb[data[b,s],d]*sqrt(D) + pos(s,d); also bf16 copy
// ---------------------------------------------------------------------------
__global__ void k_embed(const int* __restrict__ data, const float* __restrict__ emb,
                        float* __restrict__ X0, __bf16* __restrict__ X0b) {
  size_t i = (size_t)blockIdx.x * blockDim.x + threadIdx.x;
  if (i >= (size_t)B_ * S_ * D_) return;
  int d = (int)(i & (D_ - 1));
  size_t rs = i >> 10;                 // b*S + s
  int s = (int)(rs & (S_ - 1));
  int tok = data[rs];
  float f = __powf(10000.0f, -(float)(d & ~1) / (float)D_);
  float ang = (float)s * f;
  float pos = (d & 1) ? __cosf(ang) : __sinf(ang);
  float val = emb[(size_t)tok * D_ + d] * 32.0f + pos;   // sqrt(1024) = 32
  X0[i] = val;
  X0b[i] = (__bf16)val;
}

// ---------------------------------------------------------------------------
// GEMM: C[M,N] = A[M,K] (bf16, row-major) @ Bt[N,K]^T (bf16)
// 256 threads = 8 waves arranged 4(M) x 2(N); wave tile 64x32 (8 WMMAs per
// 32-deep K-step against 6 fragment loads -> ~21 flops/byte); block 256x64.
// ---------------------------------------------------------------------------
__global__ __launch_bounds__(256) void k_gemm(const __bf16* __restrict__ A,
                                              const __bf16* __restrict__ Bt,
                                              float* __restrict__ Cf,
                                              __bf16* __restrict__ Cb,
                                              int M, int N, int K) {
  int wave = threadIdx.x >> 5;
  int lane = threadIdx.x & 31;
  int wm = blockIdx.y * 256 + (wave >> 1) * 64;
  int wn = blockIdx.x * 64 + (wave & 1) * 32;
  v8f acc[4][2] = {};
  for (int k0 = 0; k0 < K; k0 += 32) {
    v16bf b0 = load_b_frag(Bt, wn, K, k0);
    v16bf b1 = load_b_frag(Bt, wn + 16, K, k0);
#pragma unroll
    for (int mi = 0; mi < 4; ++mi) {
      v16bf a = load_a_frag_clamped(A, wm + mi * 16, M, K, k0);
      acc[mi][0] = wmma_bf16(a, b0, acc[mi][0]);
      acc[mi][1] = wmma_bf16(a, b1, acc[mi][1]);
    }
  }
  // C layout: VGPR g -> row g + 8*(lane>=16), col = lane&15
  int half = lane >> 4, col = lane & 15;
#pragma unroll
  for (int mi = 0; mi < 4; ++mi) {
#pragma unroll
    for (int g = 0; g < 8; ++g) {
      int r = wm + mi * 16 + g + half * 8;
      if (r < M) {
        size_t i0 = (size_t)r * N + wn + col;
        if (Cf) { Cf[i0] = acc[mi][0][g]; Cf[i0 + 16] = acc[mi][1][g]; }
        if (Cb) { Cb[i0] = (__bf16)acc[mi][0][g]; Cb[i0 + 16] = (__bf16)acc[mi][1][g]; }
      }
    }
  }
}

// ---------------------------------------------------------------------------
// V transpose prep: Vt[b,h,d,j] = KV[b,j, HD + h*64+d], zero-padded to SkvPad
// ---------------------------------------------------------------------------
__global__ void k_vt_prep(const __bf16* __restrict__ KVb, __bf16* __restrict__ Vt,
                          int Skv, int SkvPad) {
  size_t i = (size_t)blockIdx.x * blockDim.x + threadIdx.x;
  size_t total = (size_t)B_ * H_ * DH_ * SkvPad;
  if (i >= total) return;
  int j = (int)(i % SkvPad);
  size_t r = i / SkvPad;
  int d = (int)(r % DH_);
  size_t bh = r / DH_;
  int h = (int)(bh % H_);
  int b = (int)(bh / H_);
  __bf16 v = (__bf16)0.0f;
  if (j < Skv) v = KVb[((size_t)b * Skv + j) * KV_LD + HD_ + h * DH_ + d];
  Vt[i] = v;
}

// ---------------------------------------------------------------------------
// Flash attention: one block per (q-block of 128, head, batch); 8 waves,
// each wave owns 16 query rows; 32 keys per iteration; online softmax.
// ---------------------------------------------------------------------------
__global__ __launch_bounds__(256) void k_attn(const __bf16* __restrict__ Qb,
                                              const __bf16* __restrict__ Kb,
                                              const __bf16* __restrict__ Vt,
                                              __bf16* __restrict__ Ob,
                                              int Skv, int SkvPad) {
  __shared__ __bf16 sP[8][16][32];   // per-wave P tile (C-layout -> A-layout reshape)
  int b = blockIdx.z, h = blockIdx.y;
  int wave = threadIdx.x >> 5, lane = threadIdx.x & 31;
  int half = lane >> 4, col = lane & 15;
  int m0 = blockIdx.x * 128 + wave * 16;

  const __bf16* Qbase = Qb + (size_t)b * S_ * HD_ + h * DH_;              // ld = HD_
  const __bf16* Kbase = Kb + (size_t)b * Skv * KV_LD + h * DH_;          // ld = KV_LD
  const __bf16* Vbase = Vt + (size_t)(b * H_ + h) * DH_ * SkvPad;        // ld = SkvPad

  v16bf q0 = load_a_frag(Qbase, m0, HD_, 0);
  v16bf q1 = load_a_frag(Qbase, m0, HD_, 32);

  v8f acc0 = {}, acc1 = {}, acc2 = {}, acc3 = {};
  float mr[8], lr[8];
#pragma unroll
  for (int g = 0; g < 8; ++g) { mr[g] = -1e30f; lr[g] = 0.0f; }

  for (int j0 = 0; j0 < SkvPad; j0 += 32) {
    v8f z = {};
    v8f s0 = wmma_bf16(q1, load_b_frag(Kbase, j0, KV_LD, 32),
             wmma_bf16(q0, load_b_frag(Kbase, j0, KV_LD, 0), z));
    v8f s1 = wmma_bf16(q1, load_b_frag(Kbase, j0 + 16, KV_LD, 32),
             wmma_bf16(q0, load_b_frag(Kbase, j0 + 16, KV_LD, 0), z));
    bool in0 = (j0 + col) < Skv;
    bool in1 = (j0 + 16 + col) < Skv;
#pragma unroll
    for (int g = 0; g < 8; ++g) {
      float t0 = in0 ? s0[g] * 0.125f : -1e30f;   // SCALE = 1/sqrt(64)
      float t1 = in1 ? s1[g] * 0.125f : -1e30f;
      float mx = fmaxf(t0, t1);
      mx = fmaxf(mx, __shfl_xor(mx, 1, 32));
      mx = fmaxf(mx, __shfl_xor(mx, 2, 32));
      mx = fmaxf(mx, __shfl_xor(mx, 4, 32));
      mx = fmaxf(mx, __shfl_xor(mx, 8, 32));
      float nm = fmaxf(mr[g], mx);
      float p0 = __expf(t0 - nm);
      float p1 = __expf(t1 - nm);
      float sm = p0 + p1;
      sm += __shfl_xor(sm, 1, 32);
      sm += __shfl_xor(sm, 2, 32);
      sm += __shfl_xor(sm, 4, 32);
      sm += __shfl_xor(sm, 8, 32);
      float corr = __expf(mr[g] - nm);
      lr[g] = lr[g] * corr + sm;
      mr[g] = nm;
      acc0[g] *= corr; acc1[g] *= corr; acc2[g] *= corr; acc3[g] *= corr;
      int row = g + half * 8;
      sP[wave][row][col]      = (__bf16)p0;   // wave-private; LDS in-order per wave
      sP[wave][row][col + 16] = (__bf16)p1;
    }
    v16bf pf = load_a_frag(&sP[wave][0][0], 0, 32, 0);
    acc0 = wmma_bf16(pf, load_b_frag(Vbase, 0,  SkvPad, j0), acc0);
    acc1 = wmma_bf16(pf, load_b_frag(Vbase, 16, SkvPad, j0), acc1);
    acc2 = wmma_bf16(pf, load_b_frag(Vbase, 32, SkvPad, j0), acc2);
    acc3 = wmma_bf16(pf, load_b_frag(Vbase, 48, SkvPad, j0), acc3);
  }
#pragma unroll
  for (int g = 0; g < 8; ++g) {
    float inv = 1.0f / lr[g];
    int row = m0 + g + half * 8;
    size_t base = ((size_t)b * S_ + row) * HD_ + h * DH_ + col;
    Ob[base]      = (__bf16)(acc0[g] * inv);
    Ob[base + 16] = (__bf16)(acc1[g] * inv);
    Ob[base + 32] = (__bf16)(acc2[g] * inv);
    Ob[base + 48] = (__bf16)(acc3[g] * inv);
  }
}

// ---------------------------------------------------------------------------
// out = LayerNorm(X + Y) * g + b  (one block per row of D=1024)
// ---------------------------------------------------------------------------
__global__ __launch_bounds__(256) void k_resid_ln(const float* __restrict__ X,
                                                  const float* __restrict__ Y,
                                                  const float* __restrict__ gam,
                                                  const float* __restrict__ bet,
                                                  float* __restrict__ outF,
                                                  __bf16* __restrict__ outB) {
  int row = blockIdx.x;
  const float* x = X + (size_t)row * D_;
  const float* y = Y + (size_t)row * D_;
  __shared__ float rs1[8], rs2[8];
  float v[4];
  float s1 = 0.f, s2 = 0.f;
#pragma unroll
  for (int i = 0; i < 4; ++i) {
    int d = threadIdx.x + i * 256;
    float t = x[d] + y[d];
    v[i] = t; s1 += t; s2 += t * t;
  }
#pragma unroll
  for (int m = 1; m < 32; m <<= 1) { s1 += __shfl_xor(s1, m, 32); s2 += __shfl_xor(s2, m, 32); }
  int wave = threadIdx.x >> 5;
  if ((threadIdx.x & 31) == 0) { rs1[wave] = s1; rs2[wave] = s2; }
  __syncthreads();
  float S1 = 0.f, S2 = 0.f;
#pragma unroll
  for (int w = 0; w < 8; ++w) { S1 += rs1[w]; S2 += rs2[w]; }
  float mu = S1 * (1.0f / D_);
  float var = S2 * (1.0f / D_) - mu * mu;
  float r = rsqrtf(var + 1e-5f);
#pragma unroll
  for (int i = 0; i < 4; ++i) {
    int d = threadIdx.x + i * 256;
    float o = (v[i] - mu) * r * gam[d] + bet[d];
    size_t idx = (size_t)row * D_ + d;
    if (outF) outF[idx] = o;
    if (outB) outB[idx] = (__bf16)o;
  }
}

// Build KV source for OutputAttn: Cb[b,j,:] = X1[b,1008+j] (j<16) else X1[b,j-16]
__global__ void k_build_c(const __bf16* __restrict__ X1b, __bf16* __restrict__ Cb) {
  size_t i = (size_t)blockIdx.x * blockDim.x + threadIdx.x;
  size_t total = (size_t)B_ * (S_ + DMEM_) * D_;
  if (i >= total) return;
  int d = (int)(i % D_);
  size_t r = i / D_;
  int b = (int)(r / (S_ + DMEM_));
  int j = (int)(r % (S_ + DMEM_));
  int src = (j < DMEM_) ? (S_ - DMEM_ + j) : (j - DMEM_);
  Cb[i] = X1b[((size_t)b * S_ + src) * D_ + d];
}

// new_mem = X1[:, S-16: , :]
__global__ void k_copy_mem(const float* __restrict__ X1, float* __restrict__ out) {
  size_t i = (size_t)blockIdx.x * blockDim.x + threadIdx.x;
  if (i >= (size_t)B_ * DMEM_ * D_) return;
  int d = (int)(i % D_);
  size_t r = i / D_;
  int b = (int)(r / DMEM_);
  int j = (int)(r % DMEM_);
  out[i] = X1[((size_t)b * S_ + (S_ - DMEM_) + j) * D_ + d];
}

// ---------------------------------------------------------------------------
extern "C" void kernel_launch(void* const* d_in, const int* in_sizes, int n_in,
                              void* d_out, int out_size, void* d_ws, size_t ws_size,
                              hipStream_t stream) {
  (void)in_sizes; (void)n_in; (void)out_size; (void)ws_size;
  const int*   data  = (const int*)d_in[0];
  // d_in[1] (mem) and d_in[8..15] (u_* / ffu / lnu) are dead: new_mem comes from hids
  const float* emb   = (const float*)d_in[2];
  const float* a_qw  = (const float*)d_in[3];
  const float* a_kvw = (const float*)d_in[4];
  const float* a_ow  = (const float*)d_in[5];
  const float* a_g   = (const float*)d_in[6];
  const float* a_b   = (const float*)d_in[7];
  const float* o_qw  = (const float*)d_in[16];
  const float* o_kvw = (const float*)d_in[17];
  const float* o_ow  = (const float*)d_in[18];
  const float* o_g   = (const float*)d_in[19];
  const float* o_b   = (const float*)d_in[20];
  const float* ff2_w = (const float*)d_in[21];
  const float* ln1_g = (const float*)d_in[22];
  const float* ln1_b = (const float*)d_in[23];
  const float* ln2_g = (const float*)d_in[24];
  const float* ln2_b = (const float*)d_in[25];

  const size_t M1 = (size_t)B_ * S_;     // 4096
  const int Skv3 = S_ + DMEM_;           // 1040
  const int Skv3Pad = 1056;              // next multiple of 32

  char* ws = (char*)d_ws;
  size_t off = 0;
  auto alloc = [&](size_t bytes) -> void* {
    void* p = ws + off;
    off += (bytes + 255) & ~(size_t)255;
    return p;
  };
  __bf16* wt_aqw  = (__bf16*)alloc((size_t)HD_ * D_ * 2);
  __bf16* wt_akvw = (__bf16*)alloc((size_t)2 * HD_ * D_ * 2);
  __bf16* wt_aow  = (__bf16*)alloc((size_t)D_ * HD_ * 2);
  __bf16* wt_oqw  = (__bf16*)alloc((size_t)HD_ * D_ * 2);
  __bf16* wt_okvw = (__bf16*)alloc((size_t)2 * HD_ * D_ * 2);
  __bf16* wt_oow  = (__bf16*)alloc((size_t)D_ * HD_ * 2);
  __bf16* wt_ff2  = (__bf16*)alloc((size_t)D_ * HD_ * 2);
  float*  X0   = (float*) alloc(M1 * D_ * 4);
  __bf16* X0b  = (__bf16*)alloc(M1 * D_ * 2);
  float*  X1   = (float*) alloc(M1 * D_ * 4);
  __bf16* X1b  = (__bf16*)alloc(M1 * D_ * 2);
  float*  X2   = (float*) alloc(M1 * D_ * 4);
  float*  X3   = (float*) alloc(M1 * D_ * 4);
  __bf16* X3b  = (__bf16*)alloc(M1 * D_ * 2);
  __bf16* Qb   = (__bf16*)alloc(M1 * HD_ * 2);
  __bf16* KVb  = (__bf16*)alloc((size_t)B_ * Skv3Pad * KV_LD * 2);  // slack covers padded K reads
  __bf16* Cb   = (__bf16*)alloc((size_t)B_ * Skv3 * D_ * 2);
  __bf16* Vt   = (__bf16*)alloc((size_t)B_ * H_ * DH_ * Skv3Pad * 2);
  __bf16* VECb = (__bf16*)alloc(M1 * HD_ * 2);
  float*  Of   = (float*) alloc(M1 * D_ * 4);

  dim3 blk(256);
  auto lp = [](size_t total) { return dim3((unsigned)((total + 255) / 256)); };
  auto gg = [](int M, int N) { return dim3((unsigned)(N / 64), (unsigned)((M + 255) / 256)); };

  // Weight conversion + transpose (fp32 [K,N] -> bf16 [N,K])
  k_wt_prep<<<lp((size_t)D_ * HD_),     blk, 0, stream>>>(a_qw,  wt_aqw,  D_,  HD_);
  k_wt_prep<<<lp((size_t)D_ * 2 * HD_), blk, 0, stream>>>(a_kvw, wt_akvw, D_,  2 * HD_);
  k_wt_prep<<<lp((size_t)HD_ * D_),     blk, 0, stream>>>(a_ow,  wt_aow,  HD_, D_);
  k_wt_prep<<<lp((size_t)D_ * HD_),     blk, 0, stream>>>(o_qw,  wt_oqw,  D_,  HD_);
  k_wt_prep<<<lp((size_t)D_ * 2 * HD_), blk, 0, stream>>>(o_kvw, wt_okvw, D_,  2 * HD_);
  k_wt_prep<<<lp((size_t)HD_ * D_),     blk, 0, stream>>>(o_ow,  wt_oow,  HD_, D_);
  k_wt_prep<<<lp((size_t)HD_ * D_),     blk, 0, stream>>>(ff2_w, wt_ff2,  HD_, D_);

  // Embedding + positional encoding
  k_embed<<<lp(M1 * D_), blk, 0, stream>>>(data, emb, X0, X0b);

  // ---- Self-attention (a_*) ----
  k_gemm<<<gg(4096, 1024), blk, 0, stream>>>(X0b, wt_aqw,  nullptr, Qb,  4096, 1024, 1024);
  k_gemm<<<gg(4096, 2048), blk, 0, stream>>>(X0b, wt_akvw, nullptr, KVb, 4096, 2048, 1024);
  k_vt_prep<<<lp((size_t)B_ * H_ * DH_ * S_), blk, 0, stream>>>(KVb, Vt, S_, S_);
  k_attn<<<dim3(S_ / 128, H_, B_), blk, 0, stream>>>(Qb, KVb, Vt, VECb, S_, S_);
  k_gemm<<<gg(4096, 1024), blk, 0, stream>>>(VECb, wt_aow, Of, nullptr, 4096, 1024, 1024);
  k_resid_ln<<<(unsigned)M1, blk, 0, stream>>>(X0, Of, a_g, a_b, X1, X1b);     // X1 = hids

  // new_mem output (second tuple element)
  k_copy_mem<<<lp((size_t)B_ * DMEM_ * D_), blk, 0, stream>>>(X1, (float*)d_out + M1 * D_);

  // ---- Output attention (o_*): kv over [new_mem; hids] ----
  k_build_c<<<lp((size_t)B_ * Skv3 * D_), blk, 0, stream>>>(X1b, Cb);
  k_gemm<<<gg(4096, 1024), blk, 0, stream>>>(X1b, wt_oqw, nullptr, Qb, 4096, 1024, 1024);
  k_gemm<<<gg(B_ * Skv3, 2048), blk, 0, stream>>>(Cb, wt_okvw, nullptr, KVb, B_ * Skv3, 2048, 1024);
  k_vt_prep<<<lp((size_t)B_ * H_ * DH_ * Skv3Pad), blk, 0, stream>>>(KVb, Vt, Skv3, Skv3Pad);
  k_attn<<<dim3(S_ / 128, H_, B_), blk, 0, stream>>>(Qb, KVb, Vt, VECb, Skv3, Skv3Pad);
  k_gemm<<<gg(4096, 1024), blk, 0, stream>>>(VECb, wt_oow, Of, nullptr, 4096, 1024, 1024);
  k_resid_ln<<<(unsigned)M1, blk, 0, stream>>>(X1, Of, o_g, o_b, X2, nullptr);    // attn_out
  k_resid_ln<<<(unsigned)M1, blk, 0, stream>>>(X1, X2, ln1_g, ln1_b, X3, X3b);    // out_attn

  // ---- Final FF + LN -> output ----
  k_gemm<<<gg(4096, 1024), blk, 0, stream>>>(X3b, wt_ff2, Of, nullptr, 4096, 1024, 1024);
  k_resid_ln<<<(unsigned)M1, blk, 0, stream>>>(Of, X3, ln2_g, ln2_b, (float*)d_out, nullptr);
}
